// Sae_41678362640605
// MI455X (gfx1250) — matrix operand order repaired and checked by reading, not compiled, not run on hardware
//
#include <hip/hip_runtime.h>
#include <math.h>

// ---------------- problem constants (match reference) ----------------
#define D_IN        2048
#define NUM_LATENTS 32768
#define KTOP        32
#define NTOK        4096        // B*S = 2*2048

// ---------------- encoder tiling ----------------
#define MT        32            // tokens per workgroup (two 16-row WMMA M tiles)
#define NC        512           // latents per chunk (8 waves * 64 cols)
#define KC        64            // K (d_in) chunk staged in LDS for W_enc
#define A_STRIDE  (D_IN + 8)    // bf16 elems, padded vs bank conflicts
#define B_STRIDE  (KC + 8)      // bf16 elems
#define P_STRIDE  (NC + 2)      // f32 elems

static_assert(D_IN % KC == 0, "");
static_assert(NUM_LATENTS % NC == 0, "");
static_assert(NTOK % MT == 0, "");

typedef __bf16 bf16;
typedef bf16  bf16x16 __attribute__((ext_vector_type(16)));
typedef bf16  bf16x8  __attribute__((ext_vector_type(8)));
typedef float f32x8   __attribute__((ext_vector_type(8)));

#define WMMA_BF16(A, B, C) \
    __builtin_amdgcn_wmma_f32_16x16x32_bf16(false, (A), false, (B), (short)0, (C), false, false)

// =====================================================================
// Kernel 1: fused  pre = (x - b_dec) @ W_enc^T + b_enc  ->  running top-32
// Grid: NTOK/MT blocks, 256 threads (8 wave32).
// Each wave computes a 32x64 C block as 2x4 WMMA tiles (register blocking:
// 6 fragment loads feed 8 v_wmma per K-step).
// =====================================================================
__global__ __launch_bounds__(256)
void sae_encode_topk(const float* __restrict__ x,
                     const float* __restrict__ Wenc,
                     const float* __restrict__ benc,
                     const float* __restrict__ bdec,
                     float* __restrict__ topV,
                     int*   __restrict__ topI)
{
    __shared__ bf16  As[MT * A_STRIDE];          // centered x tile, bf16, full K
    __shared__ bf16  Bs[NC * B_STRIDE];          // W_enc chunk, bf16
    __shared__ float preS[MT * P_STRIDE];        // pre-activation tile
    __shared__ float tV[MT * KTOP];              // running top-k values
    __shared__ int   tI[MT * KTOP];              // running top-k indices

    const int tid  = threadIdx.x;
    const int wave = tid >> 5;                   // 0..7 -> 64-col N sub-block
    const int lane = tid & 31;
    const int tok0 = blockIdx.x * MT;

    // ---- stage A tile once: As[m][d] = bf16(x[tok0+m][d] - b_dec[d]) ----
    for (int i = tid; i < MT * (D_IN / 4); i += 256) {
        const int row = i / (D_IN / 4);
        const int c4  = (i % (D_IN / 4)) * 4;
        float4 xv = *(const float4*)(x + (size_t)(tok0 + row) * D_IN + c4);
        float4 bv = *(const float4*)(bdec + c4);
        bf16* dst = &As[row * A_STRIDE + c4];
        dst[0] = (bf16)(xv.x - bv.x);
        dst[1] = (bf16)(xv.y - bv.y);
        dst[2] = (bf16)(xv.z - bv.z);
        dst[3] = (bf16)(xv.w - bv.w);
    }
    // ---- init top-k state ----
    for (int i = tid; i < MT * KTOP; i += 256) {
        tV[i] = -__builtin_inff();
        tI[i] = 0;
    }
    float vmin = -__builtin_inff();   // live only in owner threads tid < MT
    int   smin = 0;
    __syncthreads();

    // fragment addressing (wave32 WMMA; A and B both use contiguous k = 16*half + s)
    const int mrow = lane & 15;                  // A: lane -> M row (within tile)
    const int aks  = (lane < 16) ? 0 : 16;       // A: K sub-offset per lane half
    const int ncol = lane & 15;                  // B: lane -> N col (within tile)
    const int bks  = (lane < 16) ? 0 : 16;       // B: K offset per lane half

    for (int nb = 0; nb < NUM_LATENTS; nb += NC) {
        f32x8 acc0[4] = {{}, {}, {}, {}};        // M tile 0, N tiles 0..3
        f32x8 acc1[4] = {{}, {}, {}, {}};        // M tile 1, N tiles 0..3

        for (int kb = 0; kb < D_IN; kb += KC) {
            // ---- stage W_enc chunk: Bs[r][c] = bf16(Wenc[nb+r][kb+c]) ----
            for (int i = tid; i < NC * (KC / 4); i += 256) {
                const int r  = i / (KC / 4);
                const int c4 = (i % (KC / 4)) * 4;
                float4 wv = *(const float4*)(Wenc + (size_t)(nb + r) * D_IN + kb + c4);
                bf16* dst = &Bs[r * B_STRIDE + c4];
                dst[0] = (bf16)wv.x;
                dst[1] = (bf16)wv.y;
                dst[2] = (bf16)wv.z;
                dst[3] = (bf16)wv.w;
            }
            // ---- prefetch next K chunk (2 latent rows per thread) ----
            if (kb + KC < D_IN) {
                #pragma unroll
                for (int pr = 0; pr < NC; pr += 256) {
                    #pragma unroll
                    for (int pc = 0; pc < KC; pc += 32)
                        __builtin_prefetch(
                            Wenc + (size_t)(nb + tid + pr) * D_IN + kb + KC + pc, 0, 1);
                }
            }
            __syncthreads();

            // ---- 2x4 register-blocked WMMA over this K chunk ----
            #pragma unroll
            for (int k0 = 0; k0 < KC; k0 += 32) {
                bf16x8 a0lo = *(const bf16x8*)&As[mrow * A_STRIDE + kb + k0 + aks];
                bf16x8 a0hi = *(const bf16x8*)&As[mrow * A_STRIDE + kb + k0 + 8 + aks];
                bf16x16 a0 = __builtin_shufflevector(a0lo, a0hi,
                        0,1,2,3,4,5,6,7,8,9,10,11,12,13,14,15);
                bf16x8 a1lo = *(const bf16x8*)&As[(16 + mrow) * A_STRIDE + kb + k0 + aks];
                bf16x8 a1hi = *(const bf16x8*)&As[(16 + mrow) * A_STRIDE + kb + k0 + 8 + aks];
                bf16x16 a1 = __builtin_shufflevector(a1lo, a1hi,
                        0,1,2,3,4,5,6,7,8,9,10,11,12,13,14,15);

                #pragma unroll
                for (int t = 0; t < 4; ++t) {
                    bf16x16 b = *(const bf16x16*)
                        &Bs[(wave * 64 + t * 16 + ncol) * B_STRIDE + k0 + bks];
                    acc0[t] = WMMA_BF16(a0, b, acc0[t]);
                    acc1[t] = WMMA_BF16(a1, b, acc1[t]);
                }
            }
            __syncthreads();
        }

        // ---- spill the 32x64 pre block (+ b_enc) to LDS ----
        {
            #pragma unroll
            for (int t = 0; t < 4; ++t) {
                const int n0 = wave * 64 + t * 16 + ncol;
                const float be = benc[nb + n0];
                #pragma unroll
                for (int r = 0; r < 8; ++r) {
                    const int m = (lane < 16) ? r : (8 + r);   // C layout: VGPR r
                    preS[m        * P_STRIDE + n0] = acc0[t][r] + be;
                    preS[(16 + m) * P_STRIDE + n0] = acc1[t][r] + be;
                }
            }
        }
        __syncthreads();

        // ---- streaming replace-min top-32, one owner lane per token ----
        if (tid < MT) {
            for (int c = 0; c < NC; ++c) {
                const float v = preS[tid * P_STRIDE + c];
                if (v > vmin) {
                    tV[tid * KTOP + smin] = v;
                    tI[tid * KTOP + smin] = nb + c;
                    float mv = tV[tid * KTOP];
                    int   ms = 0;
                    #pragma unroll
                    for (int j = 1; j < KTOP; ++j) {
                        const float tv = tV[tid * KTOP + j];
                        if (tv < mv) { mv = tv; ms = j; }
                    }
                    vmin = mv; smin = ms;
                }
            }
        }
        __syncthreads();
    }

    // ---- emit top-k (unsorted; decode is a sum, order-invariant) ----
    if (tid < MT) {
        const int tok = tok0 + tid;
        for (int j = 0; j < KTOP; ++j) {
            topV[(size_t)tok * KTOP + j] = tV[tid * KTOP + j];
            topI[(size_t)tok * KTOP + j] = tI[tid * KTOP + j];
        }
    }
}

// =====================================================================
// Kernel 2: decode  x_hat[t] = b_dec + sum_k acts[t][k] * W_dec[idx[t][k]]
// Grid: NTOK blocks, 256 threads; each thread owns 8 output columns.
// =====================================================================
__global__ __launch_bounds__(256)
void sae_decode(const float* __restrict__ topV,
                const int*   __restrict__ topI,
                const float* __restrict__ Wdec,
                const float* __restrict__ bdec,
                float* __restrict__ out)
{
    const int tok = blockIdx.x;
    const int d0  = threadIdx.x * 8;

    __shared__ float av[KTOP];
    __shared__ int   ai[KTOP];
    if (threadIdx.x < KTOP) {
        av[threadIdx.x] = topV[(size_t)tok * KTOP + threadIdx.x];
        ai[threadIdx.x] = topI[(size_t)tok * KTOP + threadIdx.x];
    }
    __syncthreads();

    float4 a0 = *(const float4*)(bdec + d0);
    float4 a1 = *(const float4*)(bdec + d0 + 4);

    #pragma unroll 4
    for (int k = 0; k < KTOP; ++k) {
        const float  s  = av[k];
        const float* wr = Wdec + (size_t)ai[k] * D_IN + d0;
        float4 w0 = *(const float4*)wr;
        float4 w1 = *(const float4*)(wr + 4);
        a0.x += s * w0.x; a0.y += s * w0.y; a0.z += s * w0.z; a0.w += s * w0.w;
        a1.x += s * w1.x; a1.y += s * w1.y; a1.z += s * w1.z; a1.w += s * w1.w;
    }

    *(float4*)(out + (size_t)tok * D_IN + d0)     = a0;
    *(float4*)(out + (size_t)tok * D_IN + d0 + 4) = a1;
}

// =====================================================================
extern "C" void kernel_launch(void* const* d_in, const int* in_sizes, int n_in,
                              void* d_out, int out_size, void* d_ws, size_t ws_size,
                              hipStream_t stream)
{
    const float* x    = (const float*)d_in[0];   // [2,2048,2048]
    const float* Wenc = (const float*)d_in[1];   // [32768,2048]
    const float* benc = (const float*)d_in[2];   // [32768]
    const float* Wdec = (const float*)d_in[3];   // [32768,2048]
    const float* bdec = (const float*)d_in[4];   // [2048]
    float* out = (float*)d_out;                  // [2,2048,2048]

    float* topV = (float*)d_ws;                                        // NTOK*KTOP f32
    int*   topI = (int*)((char*)d_ws + (size_t)NTOK * KTOP * sizeof(float));

    sae_encode_topk<<<NTOK / MT, 256, 0, stream>>>(x, Wenc, benc, bdec, topV, topI);
    sae_decode<<<NTOK, 256, 0, stream>>>(topV, topI, Wdec, bdec, out);
}
